// GAT_79525614453005
// MI455X (gfx1250) — matrix operand order repaired and checked by reading, not compiled, not run on hardware
//
#include <hip/hip_runtime.h>
#include <hip/hip_bf16.h>
#include <math.h>

// ---------------- problem constants ----------------
#define N0    100000
#define N1    25000
#define N2    5000
#define E1    400000
#define E2    80000
#define INF_  256
#define HID   64
#define OUTC  47
#define HEADS 4
#define HF    256   /* HEADS*HID  */
#define HO    188   /* HEADS*OUTC */
#define HOPAD 192   /* 188 padded to 16 */
#define OPAD  48    /* 47 padded to 16 (store ld) */
#define OPADB 64    /* Wskip2t B-pad: one 4-tile N group */

typedef __attribute__((ext_vector_type(16))) __bf16 v16bf;
typedef __attribute__((ext_vector_type(8)))  float  v8f;

// float -> bf16 (round to nearest even), no reliance on __bf16 arithmetic
__device__ inline __bf16 f2bf(float f) {
  unsigned int u = __float_as_uint(f);
  unsigned int r = (u + 0x7FFFu + ((u >> 16) & 1u)) >> 16;
  unsigned short s = (unsigned short)r;
  return __builtin_bit_cast(__bf16, s);
}

// ---------------- prep kernels ----------------
__global__ void k_cast_bf16(const float* __restrict__ in, __bf16* __restrict__ out, int n) {
  int i = blockIdx.x * blockDim.x + threadIdx.x;
  if (i < n) out[i] = f2bf(in[i]);
}

// W is [Kin x Nout] row-major (f32); Wt is [NoutPad x Kin] row-major bf16, pad rows zero
__global__ void k_transpose_bf16(const float* __restrict__ W, __bf16* __restrict__ Wt,
                                 int Kin, int Nout, int NoutPad) {
  int i = blockIdx.x * blockDim.x + threadIdx.x;
  if (i >= NoutPad * Kin) return;
  int n = i / Kin, k = i % Kin;
  float v = (n < Nout) ? W[(size_t)k * Nout + n] : 0.0f;
  Wt[(size_t)n * Kin + k] = f2bf(v);
}

__global__ void k_fill_f32(float* __restrict__ p, float v, int n) {
  int i = blockIdx.x * blockDim.x + threadIdx.x;
  if (i < n) p[i] = v;
}

// ---------------- WMMA GEMM: C[MxN] = A[MxK](bf16) * Bt[NxK](bf16) + bias ----------------
// One wave computes a 16x64 strip (4 N sub-tiles): one A fragment feeds 4 WMMAs per K-step.
__global__ void __launch_bounds__(256)
k_gemm_bf16(const __bf16* __restrict__ A, const __bf16* __restrict__ Bt,
            float* __restrict__ C, int M, int K, int ldc, int Nvalid,
            int tilesM, int tilesNG,
            const float* __restrict__ bias1, const float* __restrict__ bias2) {
  int wave = blockIdx.x * 8 + (threadIdx.x >> 5);
  int lane = threadIdx.x & 31;
  if (wave >= tilesM * tilesNG) return;
  int tm = wave / tilesNG, tg = wave % tilesNG;

  int rowA  = tm * 16 + (lane & 15);  // A: lane holds row M=lane%16
  int kg    = lane >> 4;              // K-group select (contiguous 16-K halves)
  int colB0 = tg * 64 + (lane & 15);  // first of 4 B column tiles

  const __bf16* Arow = A + (size_t)rowA * K;
  const __bf16* B0 = Bt + (size_t)(colB0     ) * K;
  const __bf16* B1 = Bt + (size_t)(colB0 + 16) * K;
  const __bf16* B2 = Bt + (size_t)(colB0 + 32) * K;
  const __bf16* B3 = Bt + (size_t)(colB0 + 48) * K;

  union Frag { v16bf v; uint4 u[2]; };
  v8f acc0 = {}, acc1 = {}, acc2 = {}, acc3 = {};
  for (int kk = 0; kk < K; kk += 32) {
    Frag a, b0, b1, b2, b3;
    // A 16x32 bf16: lane group kg holds contiguous K = kk+16*kg .. +15 of its row
    const uint4* ap = reinterpret_cast<const uint4*>(Arow + kk + 16 * kg);
    a.u[0] = ap[0];
    a.u[1] = ap[1];
    __builtin_prefetch((const void*)(ap + 4), 0, 1);  // next K-chunk of this row
    // B 32x16 bf16: lane group kg holds contiguous K = kk+16*kg .. +15 of its column
    const uint4* bp;
    bp = reinterpret_cast<const uint4*>(B0 + kk + 16 * kg); b0.u[0] = bp[0]; b0.u[1] = bp[1];
    bp = reinterpret_cast<const uint4*>(B1 + kk + 16 * kg); b1.u[0] = bp[0]; b1.u[1] = bp[1];
    bp = reinterpret_cast<const uint4*>(B2 + kk + 16 * kg); b2.u[0] = bp[0]; b2.u[1] = bp[1];
    bp = reinterpret_cast<const uint4*>(B3 + kk + 16 * kg); b3.u[0] = bp[0]; b3.u[1] = bp[1];
    acc0 = __builtin_amdgcn_wmma_f32_16x16x32_bf16(false, a.v, false, b0.v, (short)0, acc0, false, false);
    acc1 = __builtin_amdgcn_wmma_f32_16x16x32_bf16(false, a.v, false, b1.v, (short)0, acc1, false, false);
    acc2 = __builtin_amdgcn_wmma_f32_16x16x32_bf16(false, a.v, false, b2.v, (short)0, acc2, false, false);
    acc3 = __builtin_amdgcn_wmma_f32_16x16x32_bf16(false, a.v, false, b3.v, (short)0, acc3, false, false);
  }

  int mg = lane >> 4;  // C/D layout: VGPR r -> M = r + 8*(lane>=16)
  int nl = lane & 15;  // N = lane%16
  v8f accs[4] = {acc0, acc1, acc2, acc3};
#pragma unroll
  for (int t = 0; t < 4; ++t) {
    int col = tg * 64 + t * 16 + nl;
    if (col >= Nvalid) continue;
    float bv = 0.0f;
    if (bias1) bv += bias1[col];
    if (bias2) bv += bias2[col];
#pragma unroll
    for (int r = 0; r < 8; ++r) {
      int m = tm * 16 + r + 8 * mg;
      if (m < M) C[(size_t)m * ldc + col] = accs[t][r] + bv;
    }
  }
}

// ---------------- attention scores: S[n,h] = sum_c H[n, h*C+c] * att[h,c] ----------------
__global__ void k_scores(const float* __restrict__ H, int ld, int C,
                         const float* __restrict__ att, float* __restrict__ S, int nNodes) {
  int t = blockIdx.x * blockDim.x + threadIdx.x;
  if (t >= nNodes * HEADS) return;
  int n = t / HEADS, h = t % HEADS;
  const float* hp = H + (size_t)n * ld + h * C;
  const float* ap = att + h * C;
  float s = 0.0f;
  for (int c = 0; c < C; ++c) s += hp[c] * ap[c];
  S[t] = s;
}

// sign-split float atomic max (works across mixed signs; init = -inf)
__device__ inline void atomicMaxF(float* addr, float v) {
  if (v >= 0.0f) atomicMax(reinterpret_cast<int*>(addr), __float_as_int(v));
  else atomicMin(reinterpret_cast<unsigned int*>(addr), (unsigned int)__float_as_int(v));
}

__global__ void k_edge_max(const int* __restrict__ esrc, const int* __restrict__ edst,
                           int E, int nLoops, const float* __restrict__ Ssrc,
                           const float* __restrict__ Sdst, float* __restrict__ amax) {
  int e = blockIdx.x * blockDim.x + threadIdx.x;
  if (e >= E + nLoops) return;
  int s = (e < E) ? esrc[e] : (e - E);
  int d = (e < E) ? edst[e] : (e - E);
#pragma unroll
  for (int h = 0; h < HEADS; ++h) {
    float a = Ssrc[s * HEADS + h] + Sdst[d * HEADS + h];
    a = (a > 0.0f) ? a : 0.2f * a;
    atomicMaxF(&amax[d * HEADS + h], a);
  }
}

__global__ void k_edge_exp(const int* __restrict__ esrc, const int* __restrict__ edst,
                           int E, int nLoops, const float* __restrict__ Ssrc,
                           const float* __restrict__ Sdst, const float* __restrict__ amax,
                           float* __restrict__ ex, float* __restrict__ denom) {
  int e = blockIdx.x * blockDim.x + threadIdx.x;
  if (e >= E + nLoops) return;
  int s = (e < E) ? esrc[e] : (e - E);
  int d = (e < E) ? edst[e] : (e - E);
#pragma unroll
  for (int h = 0; h < HEADS; ++h) {
    float a = Ssrc[s * HEADS + h] + Sdst[d * HEADS + h];
    a = (a > 0.0f) ? a : 0.2f * a;
    float ev = expf(a - amax[d * HEADS + h]);
    ex[(size_t)e * HEADS + h] = ev;
    atomicAdd(&denom[d * HEADS + h], ev);
  }
}

// one block per edge; thread c handles one feature channel
__global__ void k_edge_agg(const int* __restrict__ esrc, const int* __restrict__ edst,
                           int E, const float* __restrict__ ex, const float* __restrict__ denom,
                           const float* __restrict__ H, int ldH,
                           float* __restrict__ agg, int ldA, int nch, int chPerHead) {
  int e = blockIdx.x;
  int c = threadIdx.x;
  if (c >= nch) return;
  int s = (e < E) ? esrc[e] : (e - E);
  int d = (e < E) ? edst[e] : (e - E);
  int h = c / chPerHead;
  float coef = ex[(size_t)e * HEADS + h] / denom[d * HEADS + h];
  atomicAdd(&agg[(size_t)d * ldA + c], H[(size_t)s * ldH + c] * coef);
}

// ELU on accumulated layer-1 output, re-quantize to bf16 for layer-2 GEMMs
__global__ void k_elu_cast(const float* __restrict__ a, __bf16* __restrict__ ob, int n) {
  int i = blockIdx.x * blockDim.x + threadIdx.x;
  if (i >= n) return;
  float v = a[i];
  float r = (v > 0.0f) ? v : (expf(v) - 1.0f);
  ob[i] = f2bf(r);
}

// head-mean + skip + log_softmax over 47 classes; one wave32 per node
__global__ void __launch_bounds__(32)
k_final(const float* __restrict__ agg2, const float* __restrict__ skip2, int ldS,
        float* __restrict__ out) {
  int n = blockIdx.x;
  int lane = threadIdx.x;
  float x0 = 0.0f, x1 = 0.0f, v0 = -1e30f, v1 = -1e30f;
  if (lane < OUTC) {
    float s = 0.0f;
#pragma unroll
    for (int h = 0; h < HEADS; ++h) s += agg2[(size_t)n * HO + h * OUTC + lane];
    x0 = 0.25f * s + skip2[(size_t)n * ldS + lane];
    v0 = x0;
  }
  int c2 = lane + 32;
  if (c2 < OUTC) {
    float s = 0.0f;
#pragma unroll
    for (int h = 0; h < HEADS; ++h) s += agg2[(size_t)n * HO + h * OUTC + c2];
    x1 = 0.25f * s + skip2[(size_t)n * ldS + c2];
    v1 = x1;
  }
  float m = fmaxf(v0, v1);
  for (int off = 16; off > 0; off >>= 1) m = fmaxf(m, __shfl_xor(m, off, 32));
  float s = 0.0f;
  if (lane < OUTC) s += expf(x0 - m);
  if (c2 < OUTC) s += expf(x1 - m);
  for (int off = 16; off > 0; off >>= 1) s += __shfl_xor(s, off, 32);
  float ls = logf(s);
  if (lane < OUTC) out[(size_t)n * OUTC + lane] = x0 - m - ls;
  if (c2 < OUTC) out[(size_t)n * OUTC + c2] = x1 - m - ls;
}

// ---------------- host orchestration ----------------
extern "C" void kernel_launch(void* const* d_in, const int* in_sizes, int n_in,
                              void* d_out, int out_size, void* d_ws, size_t ws_size,
                              hipStream_t stream) {
  const float* x   = (const float*)d_in[0];
  const int*   es1 = (const int*)d_in[1];
  const int*   ed1 = (const int*)d_in[2];
  const int*   es2 = (const int*)d_in[3];
  const int*   ed2 = (const int*)d_in[4];
  const float* W1  = (const float*)d_in[5];
  const float* as1 = (const float*)d_in[6];
  const float* ad1 = (const float*)d_in[7];
  const float* b1  = (const float*)d_in[8];
  const float* Ws1 = (const float*)d_in[9];
  const float* bs1 = (const float*)d_in[10];
  const float* W2  = (const float*)d_in[11];
  const float* as2 = (const float*)d_in[12];
  const float* ad2 = (const float*)d_in[13];
  const float* b2  = (const float*)d_in[14];
  const float* Ws2 = (const float*)d_in[15];
  const float* bs2 = (const float*)d_in[16];
  float* out = (float*)d_out;

  const int MT0 = N0 / 16;         // 6250 (exact)
  const int MT1 = (N1 + 15) / 16;  // 1563
  const int MT2 = (N2 + 15) / 16;  // 313
  const int R1P = MT1 * 16;        // 25008 padded rows for layer-2 A

  char* ws = (char*)d_ws;
  size_t off = 0;
  auto alloc = [&](size_t bytes) -> char* {
    char* p = ws + off;
    off += (bytes + 255) & ~(size_t)255;
    return p;
  };
  __bf16* x_bf  = (__bf16*)alloc((size_t)N0 * INF_ * 2);
  __bf16* W1t   = (__bf16*)alloc((size_t)HF * INF_ * 2);
  __bf16* Ws1t  = (__bf16*)alloc((size_t)HF * INF_ * 2);
  __bf16* W2t   = (__bf16*)alloc((size_t)HOPAD * HF * 2);
  __bf16* Ws2t  = (__bf16*)alloc((size_t)OPADB * HF * 2);
  float*  H1    = (float*)alloc((size_t)N0 * HF * 4);
  float*  Ss1   = (float*)alloc((size_t)N0 * HEADS * 4);
  float*  Sd1   = (float*)alloc((size_t)N1 * HEADS * 4);
  float*  amax1 = (float*)alloc((size_t)N1 * HEADS * 4);
  float*  den1  = (float*)alloc((size_t)N1 * HEADS * 4);
  float*  ex1   = (float*)alloc((size_t)(E1 + N1) * HEADS * 4);
  float*  agg1  = (float*)alloc((size_t)N1 * HF * 4);
  __bf16* h1bf  = (__bf16*)alloc((size_t)R1P * HF * 2);
  float*  H2    = (float*)alloc((size_t)N1 * HOPAD * 4);
  float*  Ss2   = (float*)alloc((size_t)N1 * HEADS * 4);
  float*  Sd2   = (float*)alloc((size_t)N2 * HEADS * 4);
  float*  amax2 = (float*)alloc((size_t)N2 * HEADS * 4);
  float*  den2  = (float*)alloc((size_t)N2 * HEADS * 4);
  float*  ex2   = (float*)alloc((size_t)(E2 + N2) * HEADS * 4);
  float*  agg2  = (float*)alloc((size_t)N2 * HO * 4);
  float*  skip2 = (float*)alloc((size_t)N2 * OPAD * 4);
  (void)ws_size; (void)in_sizes; (void)n_in; (void)out_size;

  const int T = 256;
  // --- prep: bf16 casts / weight transposes ---
  k_cast_bf16<<<(N0 * INF_ + T - 1) / T, T, 0, stream>>>(x, x_bf, N0 * INF_);
  k_transpose_bf16<<<(HF * INF_ + T - 1) / T, T, 0, stream>>>(W1, W1t, INF_, HF, HF);
  k_transpose_bf16<<<(HF * INF_ + T - 1) / T, T, 0, stream>>>(Ws1, Ws1t, INF_, HF, HF);
  k_transpose_bf16<<<(HOPAD * HF + T - 1) / T, T, 0, stream>>>(W2, W2t, HF, HO, HOPAD);
  k_transpose_bf16<<<(OPADB * HF + T - 1) / T, T, 0, stream>>>(Ws2, Ws2t, HF, OUTC, OPADB);

  // --- layer 1 ---
  int tiles = MT0 * 4;  // H1 = x @ W1 (256 cols = 4 N-groups of 64)
  k_gemm_bf16<<<(tiles + 7) / 8, 256, 0, stream>>>(x_bf, W1t, H1, N0, INF_, HF, HF,
                                                   MT0, 4, nullptr, nullptr);
  k_scores<<<(N0 * HEADS + T - 1) / T, T, 0, stream>>>(H1, HF, HID, as1, Ss1, N0);
  k_scores<<<(N1 * HEADS + T - 1) / T, T, 0, stream>>>(H1, HF, HID, ad1, Sd1, N1);
  k_fill_f32<<<(N1 * HEADS + T - 1) / T, T, 0, stream>>>(amax1, -INFINITY, N1 * HEADS);
  k_fill_f32<<<(N1 * HEADS + T - 1) / T, T, 0, stream>>>(den1, 0.0f, N1 * HEADS);
  tiles = MT1 * 4;  // agg1 = x_t @ Wskip1 + b1 + bskip1 (aggregation accumulates on top)
  k_gemm_bf16<<<(tiles + 7) / 8, 256, 0, stream>>>(x_bf, Ws1t, agg1, N1, INF_, HF, HF,
                                                   MT1, 4, b1, bs1);
  int nE1 = E1 + N1;
  k_edge_max<<<(nE1 + T - 1) / T, T, 0, stream>>>(es1, ed1, E1, N1, Ss1, Sd1, amax1);
  k_edge_exp<<<(nE1 + T - 1) / T, T, 0, stream>>>(es1, ed1, E1, N1, Ss1, Sd1, amax1, ex1, den1);
  k_edge_agg<<<nE1, 256, 0, stream>>>(es1, ed1, E1, ex1, den1, H1, HF, agg1, HF, HF, HID);
  k_elu_cast<<<(N1 * HF + T - 1) / T, T, 0, stream>>>(agg1, h1bf, N1 * HF);

  // --- layer 2 ---
  tiles = MT1 * 3;  // H2 = h @ W2 (192 cols = 3 N-groups)
  k_gemm_bf16<<<(tiles + 7) / 8, 256, 0, stream>>>(h1bf, W2t, H2, N1, INF_, HOPAD, HO,
                                                   MT1, 3, nullptr, nullptr);
  k_scores<<<(N1 * HEADS + T - 1) / T, T, 0, stream>>>(H2, HOPAD, OUTC, as2, Ss2, N1);
  k_scores<<<(N2 * HEADS + T - 1) / T, T, 0, stream>>>(H2, HOPAD, OUTC, ad2, Sd2, N2);
  k_fill_f32<<<(N2 * HEADS + T - 1) / T, T, 0, stream>>>(amax2, -INFINITY, N2 * HEADS);
  k_fill_f32<<<(N2 * HEADS + T - 1) / T, T, 0, stream>>>(den2, 0.0f, N2 * HEADS);
  k_fill_f32<<<(N2 * HO + T - 1) / T, T, 0, stream>>>(agg2, 0.0f, N2 * HO);
  tiles = MT2 * 1;  // skip2 = x_t2 @ Wskip2 + b2 + bskip2 (64-col padded B, 1 group)
  k_gemm_bf16<<<(tiles + 7) / 8, 256, 0, stream>>>(h1bf, Ws2t, skip2, N2, INF_, OPAD, OUTC,
                                                   MT2, 1, b2, bs2);
  int nE2 = E2 + N2;
  k_edge_max<<<(nE2 + T - 1) / T, T, 0, stream>>>(es2, ed2, E2, N2, Ss2, Sd2, amax2);
  k_edge_exp<<<(nE2 + T - 1) / T, T, 0, stream>>>(es2, ed2, E2, N2, Ss2, Sd2, amax2, ex2, den2);
  k_edge_agg<<<nE2, HOPAD, 0, stream>>>(es2, ed2, E2, ex2, den2, H2, HOPAD, agg2, HO, HO, OUTC);
  k_final<<<N2, 32, 0, stream>>>(agg2, skip2, OPAD, out);
}
